// EMG_Classifier_25022479466721
// MI455X (gfx1250) — matrix operand order repaired and checked by reading, not compiled, not run on hardware
//
#include <hip/hip_runtime.h>

// ---------------------------------------------------------------------------
// Problem constants (match reference)
// ---------------------------------------------------------------------------
#define N_NODES   20000
#define DEG       32
#define FDIM      128
#define G4        512          // 4 * FDIM gates
#define N_LAYERS  6
#define N_GRAPHS  64
#define N_CLASSES 10

#define TM        64           // nodes per workgroup tile
#define WPAD      136          // padded row stride (bf16 elems) -> 272B, bank-conflict free

typedef __attribute__((ext_vector_type(16))) __bf16        v16bf;
typedef __attribute__((ext_vector_type(8)))  float         v8f;
typedef __attribute__((ext_vector_type(4)))  float         f32x4;
typedef __attribute__((ext_vector_type(4)))  unsigned int  u32x4;
typedef __attribute__((ext_vector_type(2)))  unsigned int  u32x2;

__device__ __forceinline__ unsigned short f2bf(float x) {
    unsigned int u = __float_as_uint(x);
    u += 0x7FFFu + ((u >> 16) & 1u);          // round-to-nearest-even
    return (unsigned short)(u >> 16);
}
__device__ __forceinline__ float bf2f(unsigned short s) {
    return __uint_as_float((unsigned int)s << 16);
}
__device__ __forceinline__ unsigned int pack2(float a, float b) {
    return (unsigned int)f2bf(a) | ((unsigned int)f2bf(b) << 16);
}
__device__ __forceinline__ float sigf(float x) {
    return 0.5f * tanhf(0.5f * x) + 0.5f;     // v_tanh_f32 transcendental path
}
__device__ __forceinline__ v8f splat8(float x) {
    v8f r;
#pragma unroll
    for (int i = 0; i < 8; ++i) r[i] = x;
    return r;
}
__device__ __forceinline__ v8f wmma_bf16(v16bf a, v16bf b, v8f c) {
    return __builtin_amdgcn_wmma_f32_16x16x32_bf16(
        /*neg_a=*/false, a, /*neg_b=*/false, b,
        /*c_mod=*/(short)0, c, /*reuse_a=*/false, /*reuse_b=*/false);
}

union FragU { u32x4 u[2]; unsigned short s[16]; v16bf v; };

// A fragment (16x32 bf16, MxK) from padded LDS: lane L -> row M = rowBase+(L&15);
//   K(j) = k0 + (j%8) + 8*(2*(j/8) + (L>>4))  => two 16B LDS reads.
__device__ __forceinline__ v16bf loadA(const unsigned short* base, int rowBase,
                                       int k0, int nrow, int khi) {
    FragU f;
    const unsigned short* p = base + (rowBase + nrow) * WPAD + k0 + 8 * khi;
    f.u[0] = *reinterpret_cast<const u32x4*>(p);
    f.u[1] = *reinterpret_cast<const u32x4*>(p + 16);
    return f.v;
}
// B fragment (32x16 bf16, KxN) built once from GLOBAL f32 weights:
//   lane L -> col N folded into wrow; K(j) = k0 + (j%8) + 8*(2*(j/8) + (L>>4)).
__device__ __forceinline__ v16bf loadBglobal(const float* __restrict__ wrow,
                                             int k0, int khi) {
    FragU f;
    const float* p = wrow + k0 + 8 * khi;
#pragma unroll
    for (int j = 0; j < 16; ++j) f.s[j] = f2bf(p[j < 8 ? j : j + 8]);
    return f.v;
}

// ---------------------------------------------------------------------------
// One SAGEConv-LSTM layer: h_out = fc_self(h_in) + fc_neigh(LSTM(h_in[neigh]))
// Inter-layer h is bf16 (row = 256B), so the per-step neighbor gather is a
// pure b128 global->LDS copy. Wave w owns feature columns [16w,16w+16) for ALL
// 64 nodes; its 32 B fragments (Wih+Whh, 4 gates, K=128) stay in registers for
// the whole 32-step scan. h_s is double-buffered in LDS (no intra-step races).
// ---------------------------------------------------------------------------
__global__ void __launch_bounds__(256)
sage_lstm_layer_kernel(const unsigned short* __restrict__ h_in,
                       const int*   __restrict__ neigh,
                       const float* __restrict__ Wih,  const float* __restrict__ Whh,
                       const float* __restrict__ bih,  const float* __restrict__ bhh,
                       const float* __restrict__ Wself,const float* __restrict__ bself,
                       const float* __restrict__ Wnei, const float* __restrict__ bnei,
                       unsigned short* __restrict__ h_out, int relu_out)
{
    __shared__ unsigned short ldsX[TM * WPAD];        // gathered neighbor feats (bf16)
    __shared__ unsigned short ldsH[2][TM * WPAD];     // double-buffered hidden state
    __shared__ int            ldsIdx[TM];

    const int tid  = threadIdx.x;
    const int lane = tid & 31;
    const int wave = tid >> 5;
    const int nrow = lane & 15;
    const int khi  = lane >> 4;
    const int colN = wave * 16 + nrow;     // this lane's feature column (0..127)
    const int wgBase = blockIdx.x * TM;

    // ---- register-resident B fragments: Bw[matrix][gate][kchunk] ----------
    v16bf Bw[2][4][4];
#pragma unroll
    for (int m = 0; m < 2; ++m) {
        const float* Wm = (m == 0) ? Wih : Whh;
#pragma unroll
        for (int g = 0; g < 4; ++g) {
            const float* wrow = Wm + (size_t)(g * FDIM + colN) * FDIM;
#pragma unroll
            for (int kk = 0; kk < 4; ++kk)
                Bw[m][g][kk] = loadBglobal(wrow, kk * 32, khi);
        }
    }
    // per-lane gate biases (constant over scan)
    const float bI = bih[0 * FDIM + colN] + bhh[0 * FDIM + colN];
    const float bF = bih[1 * FDIM + colN] + bhh[1 * FDIM + colN];
    const float bG = bih[2 * FDIM + colN] + bhh[2 * FDIM + colN];
    const float bO = bih[3 * FDIM + colN] + bhh[3 * FDIM + colN];

    // h_s buffer 0 = 0
    for (int q = tid; q < TM * WPAD / 2; q += 256)
        reinterpret_cast<unsigned int*>(ldsH[0])[q] = 0u;

    float cs[4][8];                         // c_s: 64 nodes x 16 cols / wave
#pragma unroll
    for (int a = 0; a < 4; ++a)
#pragma unroll
        for (int b = 0; b < 8; ++b) cs[a][b] = 0.f;

    // ------------------------- LSTM scan over DEG steps --------------------
    for (int t = 0; t < DEG; ++t) {
        __syncthreads();                    // prior-step ldsX/ldsIdx reads complete
        if (tid < TM) {
            const int node = wgBase + tid;
            ldsIdx[tid] = (node < N_NODES) ? neigh[node * DEG + t] : 0;
        }
        __syncthreads();
        // gather x_t = h_in[neighbors[:, t]]  -- pure b128 copy (bf16 rows)
        for (int q = tid; q < TM * (FDIM / 8); q += 256) {     // 16 chunks/row
            const int r = q >> 4, c = q & 15;
            const u32x4 v = reinterpret_cast<const u32x4*>(
                h_in + (size_t)ldsIdx[r] * FDIM)[c];
            *reinterpret_cast<u32x4*>(&ldsX[r * WPAD + c * 8]) = v;
        }
        __syncthreads();                    // X visible; prev h_s writes visible

        const unsigned short* Hr = ldsH[t & 1];
        unsigned short*       Hw = ldsH[(t + 1) & 1];

#pragma unroll
        for (int mt = 0; mt < 4; ++mt) {    // 16-node sub-tiles
            v16bf aX[4], aH[4];
#pragma unroll
            for (int kk = 0; kk < 4; ++kk) {
                aX[kk] = loadA(ldsX, mt * 16, kk * 32, nrow, khi);
                aH[kk] = loadA(Hr,  mt * 16, kk * 32, nrow, khi);
            }
            v8f ai = splat8(0.f), af = splat8(0.f), ag = splat8(0.f), ao = splat8(0.f);
#pragma unroll
            for (int kk = 0; kk < 4; ++kk) {          // x_t @ Wih^T (B in regs)
                ai = wmma_bf16(aX[kk], Bw[0][0][kk], ai);
                af = wmma_bf16(aX[kk], Bw[0][1][kk], af);
                ag = wmma_bf16(aX[kk], Bw[0][2][kk], ag);
                ao = wmma_bf16(aX[kk], Bw[0][3][kk], ao);
            }
#pragma unroll
            for (int kk = 0; kk < 4; ++kk) {          // h_s @ Whh^T (B in regs)
                ai = wmma_bf16(aH[kk], Bw[1][0][kk], ai);
                af = wmma_bf16(aH[kk], Bw[1][1][kk], af);
                ag = wmma_bf16(aH[kk], Bw[1][2][kk], ag);
                ao = wmma_bf16(aH[kk], Bw[1][3][kk], ao);
            }
            // LSTM cell update (torch gate order i,f,g,o)
#pragma unroll
            for (int v = 0; v < 8; ++v) {
                const float iv = sigf(ai[v] + bI);
                const float fv = sigf(af[v] + bF);
                const float gv = tanhf(ag[v] + bG);
                const float ov = sigf(ao[v] + bO);
                const float c  = fv * cs[mt][v] + iv * gv;
                cs[mt][v] = c;
                Hw[(mt * 16 + v + 8 * khi) * WPAD + colN] = f2bf(ov * tanhf(c));
            }
        }
    }

    // -------- output: fc_self(h_in) + fc_neigh(h_n) (+ ReLU) ---------------
    __syncthreads();                        // scan reads of ldsX complete
    for (int q = tid; q < TM * (FDIM / 8); q += 256) {   // self features (copy)
        const int r = q >> 4, c = q & 15;
        const int node = wgBase + r;
        const u32x4 v = reinterpret_cast<const u32x4*>(
            h_in + (size_t)(node < N_NODES ? node : 0) * FDIM)[c];
        *reinterpret_cast<u32x4*>(&ldsX[r * WPAD + c * 8]) = v;
    }
    __syncthreads();

    const unsigned short* Hf = ldsH[DEG & 1];            // final hidden state
    v16bf Bs[2][4];
#pragma unroll
    for (int kk = 0; kk < 4; ++kk) {
        Bs[0][kk] = loadBglobal(Wself + (size_t)colN * FDIM, kk * 32, khi);
        Bs[1][kk] = loadBglobal(Wnei  + (size_t)colN * FDIM, kk * 32, khi);
    }
    const float bOut = bself[colN] + bnei[colN];

#pragma unroll
    for (int mt = 0; mt < 4; ++mt) {
        v8f acc = splat8(0.f);
#pragma unroll
        for (int kk = 0; kk < 4; ++kk)
            acc = wmma_bf16(loadA(ldsX, mt * 16, kk * 32, nrow, khi), Bs[0][kk], acc);
#pragma unroll
        for (int kk = 0; kk < 4; ++kk)
            acc = wmma_bf16(loadA(Hf,  mt * 16, kk * 32, nrow, khi), Bs[1][kk], acc);
#pragma unroll
        for (int v = 0; v < 8; ++v) {
            float x = acc[v] + bOut;
            if (relu_out) x = fmaxf(x, 0.f);
            const int node = wgBase + mt * 16 + v + 8 * khi;
            if (node < N_NODES) h_out[(size_t)node * FDIM + colN] = f2bf(x);
        }
    }
}

// ---------------------------------------------------------------------------
// One-time f32 -> bf16 conversion of the input features
// ---------------------------------------------------------------------------
__global__ void cvt_kernel(const float* __restrict__ in,
                           unsigned short* __restrict__ outb, int n4) {
    const int i = blockIdx.x * blockDim.x + threadIdx.x;
    if (i < n4) {
        const f32x4 v = reinterpret_cast<const f32x4*>(in)[i];
        u32x2 p; p[0] = pack2(v[0], v[1]); p[1] = pack2(v[2], v[3]);
        reinterpret_cast<u32x2*>(outb)[i] = p;
    }
}

// ---------------------------------------------------------------------------
// Pooling + MLP head (negligible cost)
// ---------------------------------------------------------------------------
__global__ void zero_kernel(float* sums, float* cnts) {
    const int i = blockIdx.x * blockDim.x + threadIdx.x;
    if (i < N_GRAPHS * FDIM) sums[i] = 0.f;
    if (i < N_GRAPHS) cnts[i] = 0.f;
}

__global__ void pool_kernel(const unsigned short* __restrict__ h,
                            const int* __restrict__ gid,
                            float* sums, float* cnts) {
    const int t = blockIdx.x * blockDim.x + threadIdx.x;   // 8 threads per node
    if (t >= N_NODES * 8) return;
    const int node = t >> 3, seg = t & 7;
    const int g = gid[node];
    const unsigned short* hp = h + (size_t)node * FDIM + seg * 16;
    float* sp = sums + g * FDIM + seg * 16;
#pragma unroll
    for (int k = 0; k < 16; ++k) atomicAdd(&sp[k], bf2f(hp[k]));
    if (seg == 0) atomicAdd(&cnts[g], 1.0f);
}

__global__ void __launch_bounds__(256)
head_kernel(const float* __restrict__ sums, const float* __restrict__ cnts,
            const float* __restrict__ mlpW, const float* __restrict__ mlpb,
            const float* __restrict__ Wout, const float* __restrict__ bout,
            float* __restrict__ out) {
    __shared__ float gbuf[N_GRAPHS * FDIM];
    __shared__ float tbuf[N_GRAPHS * FDIM];
    const int tid = threadIdx.x;
    for (int i = tid; i < N_GRAPHS * FDIM; i += 256)
        gbuf[i] = sums[i] / fmaxf(cnts[i >> 7], 1.0f);
    __syncthreads();
    for (int layer = 0; layer < 3; ++layer) {
        const float* W = mlpW + (size_t)layer * FDIM * FDIM;
        const float* b = mlpb + layer * FDIM;
        for (int i = tid; i < N_GRAPHS * FDIM; i += 256) {
            const int r = i >> 7, c = i & 127;
            float acc = b[c];
            for (int k = 0; k < FDIM; ++k) acc += gbuf[r * FDIM + k] * W[c * FDIM + k];
            tbuf[i] = (layer < 2) ? fmaxf(acc, 0.f) : acc;
        }
        __syncthreads();
        for (int i = tid; i < N_GRAPHS * FDIM; i += 256) gbuf[i] = tbuf[i];
        __syncthreads();
    }
    for (int i = tid; i < N_GRAPHS * N_CLASSES; i += 256) {
        const int r = i / N_CLASSES, c = i % N_CLASSES;
        float acc = bout[c];
        for (int k = 0; k < FDIM; ++k) acc += gbuf[r * FDIM + k] * Wout[c * FDIM + k];
        out[i] = acc;
    }
}

// ---------------------------------------------------------------------------
extern "C" void kernel_launch(void* const* d_in, const int* in_sizes, int n_in,
                              void* d_out, int out_size, void* d_ws, size_t ws_size,
                              hipStream_t stream) {
    const float* h0     = (const float*)d_in[0];
    const int*   neigh  = (const int*)  d_in[1];
    const int*   gid    = (const int*)  d_in[2];
    const float* Wih    = (const float*)d_in[3];
    const float* Whh    = (const float*)d_in[4];
    const float* bih    = (const float*)d_in[5];
    const float* bhh    = (const float*)d_in[6];
    const float* Wself  = (const float*)d_in[7];
    const float* bself  = (const float*)d_in[8];
    const float* Wnei   = (const float*)d_in[9];
    const float* bnei   = (const float*)d_in[10];
    const float* mlpW   = (const float*)d_in[11];
    const float* mlpb   = (const float*)d_in[12];
    const float* Wout   = (const float*)d_in[13];
    const float* bout   = (const float*)d_in[14];
    float* out = (float*)d_out;

    unsigned short* hA = (unsigned short*)d_ws;                 // bf16 h buffers
    unsigned short* hB = hA + (size_t)N_NODES * FDIM;
    float* sums = (float*)(hB + (size_t)N_NODES * FDIM);
    float* cnts = sums + N_GRAPHS * FDIM;

    // one-time: input features f32 -> bf16
    const int n4 = N_NODES * FDIM / 4;
    cvt_kernel<<<(n4 + 255) / 256, 256, 0, stream>>>(h0, hA, n4);

    const int nblk = (N_NODES + TM - 1) / TM;
    const unsigned short* src = hA;
    unsigned short* dst = hB;
    for (int l = 0; l < N_LAYERS; ++l) {
        sage_lstm_layer_kernel<<<nblk, 256, 0, stream>>>(
            src, neigh,
            Wih + (size_t)l * G4 * FDIM, Whh + (size_t)l * G4 * FDIM,
            bih + l * G4, bhh + l * G4,
            Wself + (size_t)l * FDIM * FDIM, bself + l * FDIM,
            Wnei  + (size_t)l * FDIM * FDIM, bnei  + l * FDIM,
            dst, (l < N_LAYERS - 1) ? 1 : 0);
        const unsigned short* nsrc = dst;
        dst = (dst == hA) ? hB : hA;
        src = nsrc;
    }
    zero_kernel<<<(N_GRAPHS * FDIM + 255) / 256, 256, 0, stream>>>(sums, cnts);
    pool_kernel<<<(N_NODES * 8 + 255) / 256, 256, 0, stream>>>(src, gid, sums, cnts);
    head_kernel<<<1, 256, 0, stream>>>(sums, cnts, mlpW, mlpb, Wout, bout, out);
}
